// DecoderBlock_68496138436735
// MI455X (gfx1250) — matrix operand order repaired and checked by reading, not compiled, not run on hardware
//
#include <hip/hip_runtime.h>

// ---------------------------------------------------------------------------
// Decoder block (pre-LN GPT style) for MI455X / gfx1250, wave32 + WMMA bf16.
//   B=2 S=2048 E=768 H=12 D=64, M = B*S = 4096.
// v3: GEMM K-loop is double-buffered (LDS ping-pong, one barrier/iter) so
// global b128 loads of the next tile overlap the 64 WMMAs of the current
// tile. All operands bf16 end-to-end (weights converted once per call, LN /
// GEMM epilogues emit bf16); hot loops do zero format conversions.
// ---------------------------------------------------------------------------

#define BB 2
#define SS 2048
#define EE 768
#define HH 12
#define DD 64
#define E3 (3 * EE)
#define MM (BB * SS)

typedef __attribute__((ext_vector_type(16))) __bf16 v16bf;
typedef __attribute__((ext_vector_type(8))) float v8f;
typedef __attribute__((vector_size(16))) unsigned int uint32x4;   // trivial, union-safe
typedef __attribute__((vector_size(16))) float floatx4;
typedef __attribute__((vector_size(8)))  unsigned short ushortx4;

union Frag {
  v16bf v;
  unsigned short u[16];
  uint32x4 q[2];
};

union Q16 {
  uint32x4 q;
  unsigned short u[8];
};

__device__ __forceinline__ unsigned short f2bf16(float f) {
  unsigned int u = __float_as_uint(f);
  u += 0x7fffu + ((u >> 16) & 1u);  // round-to-nearest-even
  return (unsigned short)(u >> 16);
}

__device__ __forceinline__ v8f vzero8() {
  v8f z;
#pragma unroll
  for (int i = 0; i < 8; ++i) z[i] = 0.0f;
  return z;
}

// ---------------------------------------------------------------------------
// One-shot f32 -> bf16 tensor conversion (weights), vectorized 4-wide.
// ---------------------------------------------------------------------------
__global__ __launch_bounds__(256) void cvt_bf16_kernel(const float* __restrict__ in,
                                                       unsigned short* __restrict__ out,
                                                       int n4) {
  int i = blockIdx.x * 256 + threadIdx.x;
  if (i < n4) {
    floatx4 f = *reinterpret_cast<const floatx4*>(in + (size_t)i * 4);
    ushortx4 u;
#pragma unroll
    for (int j = 0; j < 4; ++j) u[j] = f2bf16(f[j]);
    *reinterpret_cast<ushortx4*>(out + (size_t)i * 4) = u;
  }
}

// ---------------------------------------------------------------------------
// LayerNorm: one 256-thread workgroup per row of E=768; bf16 output.
// ---------------------------------------------------------------------------
__global__ __launch_bounds__(256) void ln_kernel(const float* __restrict__ x,
                                                 const float* __restrict__ g,
                                                 const float* __restrict__ b,
                                                 unsigned short* __restrict__ out) {
  __shared__ float red[256];
  const int t = threadIdx.x;
  const size_t row = blockIdx.x;
  const float* xr = x + row * EE;
  float v[3];
  float s = 0.f;
#pragma unroll
  for (int i = 0; i < 3; ++i) {
    v[i] = xr[t + 256 * i];
    s += v[i];
  }
  red[t] = s;
  __syncthreads();
  for (int off = 128; off > 0; off >>= 1) {
    if (t < off) red[t] += red[t + off];
    __syncthreads();
  }
  const float mu = red[0] * (1.0f / EE);
  __syncthreads();
  float s2 = 0.f;
#pragma unroll
  for (int i = 0; i < 3; ++i) {
    float d = v[i] - mu;
    s2 += d * d;
  }
  red[t] = s2;
  __syncthreads();
  for (int off = 128; off > 0; off >>= 1) {
    if (t < off) red[t] += red[t + off];
    __syncthreads();
  }
  const float rstd = rsqrtf(red[0] * (1.0f / EE) + 1e-5f);
  unsigned short* orow = out + row * EE;
#pragma unroll
  for (int i = 0; i < 3; ++i) {
    int c = t + 256 * i;
    orow[c] = f2bf16((v[i] - mu) * rstd * g[c] + b[c]);
  }
}

// ---------------------------------------------------------------------------
// GEMM: C[M,N] = act(A[M,K] @ W[K,N] + bias) (+ residual)
//   A, W: bf16.  OUTBF ? bf16 output : f32 output (+f32 residual).
// Block tile 128x128, BK=32, 256 threads = 8 waves (4x2), wave tile 32x64.
// Double-buffered LDS ping-pong: next tile's global loads overlap the
// current tile's 64 WMMAs; one __syncthreads per K step.
// ---------------------------------------------------------------------------
template <int ACT, bool RESID, bool OUTBF>
__global__ __launch_bounds__(256) void gemm_bf16_kernel(
    const unsigned short* __restrict__ A, const unsigned short* __restrict__ W,
    const float* __restrict__ bias, const float* __restrict__ R,
    void* __restrict__ Cout, int Mdim, int Ndim, int Kdim) {
  __shared__ alignas(16) unsigned short As[2][128][40];  // [m][k] bf16, 80B row stride
  __shared__ alignas(16) unsigned short Bs[2][128][40];  // [n][k] bf16 (transposed W)

  const int tid = threadIdx.x;
  const int lane = tid & 31;
  const int wave = tid >> 5;
  const int lh = lane & 15;
  const int kg = lane >> 4;
  const int wm = wave >> 1;  // 0..3
  const int wn = wave & 1;   // 0..1
  const int m0 = blockIdx.y * 128;
  const int n0 = blockIdx.x * 128;

  v8f acc[2][4];
#pragma unroll
  for (int fm = 0; fm < 2; ++fm)
#pragma unroll
    for (int fn = 0; fn < 4; ++fn) acc[fm][fn] = vzero8();

  uint32x4 ra[2];   // register staging for A tile (2 x b128 per thread)
  ushortx4 rw[4];   // register staging for W tile (4 x b64 per thread)

  auto load_global = [&](int kk) {
#pragma unroll
    for (int i = 0; i < 2; ++i) {
      int idx = tid + 256 * i;
      int row = idx >> 2;
      int col = (idx & 3) * 8;
      ra[i] = *reinterpret_cast<const uint32x4*>(A + (size_t)(m0 + row) * Kdim + kk + col);
    }
#pragma unroll
    for (int i = 0; i < 4; ++i) {
      int idx = tid + 256 * i;
      int kr = idx >> 5;
      int col = (idx & 31) * 4;
      rw[i] = *reinterpret_cast<const ushortx4*>(W + (size_t)(kk + kr) * Ndim + n0 + col);
    }
  };
  auto store_lds = [&](int buf) {
#pragma unroll
    for (int i = 0; i < 2; ++i) {
      int idx = tid + 256 * i;
      int row = idx >> 2;
      int col = (idx & 3) * 8;
      *reinterpret_cast<uint32x4*>(&As[buf][row][col]) = ra[i];
    }
#pragma unroll
    for (int i = 0; i < 4; ++i) {
      int idx = tid + 256 * i;
      int kr = idx >> 5;
      int col = (idx & 31) * 4;
      Bs[buf][col + 0][kr] = rw[i][0];
      Bs[buf][col + 1][kr] = rw[i][1];
      Bs[buf][col + 2][kr] = rw[i][2];
      Bs[buf][col + 3][kr] = rw[i][3];
    }
  };

  // prologue: stage tile 0
  load_global(0);
  store_lds(0);
  __syncthreads();

  int buf = 0;
  for (int kk = 0; kk < Kdim; kk += 32) {
    const bool more = (kk + 32) < Kdim;
    if (more) {
      load_global(kk + 32);  // overlap with WMMA below
      __builtin_prefetch(A + (size_t)(m0 + (tid & 127)) * Kdim + kk + 64, 0, 1);
      __builtin_prefetch(W + (size_t)(kk + 64 + (tid >> 3)) * Ndim + n0 + (tid & 7) * 16, 0, 1);
    }

    // ---- compute current tile: b128 ds fragment loads + 8 WMMA ----
    Frag af[2], bfr[4];
#pragma unroll
    for (int fm = 0; fm < 2; ++fm) {
      int m = wm * 32 + fm * 16 + lh;
      // A 16x32 bf16 layout: lane half kg holds K in {8kg..8kg+7, 16+8kg..23+8kg}
      af[fm].q[0] = *reinterpret_cast<const uint32x4*>(&As[buf][m][8 * kg]);
      af[fm].q[1] = *reinterpret_cast<const uint32x4*>(&As[buf][m][16 + 8 * kg]);
    }
#pragma unroll
    for (int fn = 0; fn < 4; ++fn) {
      int n = wn * 64 + fn * 16 + lh;
      // B 32x16 bf16 layout: lane half kg holds K in {8kg..8kg+7, 16+8kg..23+8kg}
      bfr[fn].q[0] = *reinterpret_cast<const uint32x4*>(&Bs[buf][n][8 * kg]);
      bfr[fn].q[1] = *reinterpret_cast<const uint32x4*>(&Bs[buf][n][16 + 8 * kg]);
    }
#pragma unroll
    for (int fm = 0; fm < 2; ++fm)
#pragma unroll
      for (int fn = 0; fn < 4; ++fn)
        acc[fm][fn] = __builtin_amdgcn_wmma_f32_16x16x32_bf16(
            false, af[fm].v, false, bfr[fn].v, (short)0, acc[fm][fn], false, false);

    if (more) store_lds(buf ^ 1);  // other buffer: no hazard with this tile's reads
    __syncthreads();
    buf ^= 1;
  }

  // ---- epilogue: bias, activation, residual, store ----
#pragma unroll
  for (int fm = 0; fm < 2; ++fm)
#pragma unroll
    for (int fn = 0; fn < 4; ++fn) {
      int n = n0 + wn * 64 + fn * 16 + lh;
      float bn = bias[n];
#pragma unroll
      for (int i = 0; i < 8; ++i) {
        int m = m0 + wm * 32 + fm * 16 + i + 8 * kg;  // C layout: VGPR i -> row i (+8 hi half)
        float vv = acc[fm][fn][i] + bn;
        if (ACT == 1) vv = 0.5f * vv * (1.0f + erff(vv * 0.70710678118f));
        size_t o = (size_t)m * Ndim + n;
        if constexpr (OUTBF) {
          reinterpret_cast<unsigned short*>(Cout)[o] = f2bf16(vv);
        } else {
          if (RESID) vv += R[o];
          reinterpret_cast<float*>(Cout)[o] = vv;
        }
      }
    }
}

// ---------------------------------------------------------------------------
// Causal attention (no 1/sqrt(D) scaling, per reference). All bf16 inputs.
// qkv16: [B,S,3E] bf16. out16: [B,S,E] bf16 (heads re-merged).
// One wave per 16-query tile; 4 waves/block; kv tiles of 32 keys.
// Q/K fragments: direct b128 global loads (contiguous along d).
// V: staged transposed ([d][key]) in per-wave LDS -> b128 ds fragment loads.
// P relayout (C-layout f32 -> A-layout bf16) via per-wave LDS.
// ---------------------------------------------------------------------------
__global__ __launch_bounds__(128) void attn_kernel(const unsigned short* __restrict__ qkv16,
                                                   unsigned short* __restrict__ out16) {
  __shared__ alignas(16) unsigned short Vst[4][64][32];  // per-wave V^T tile
  __shared__ alignas(16) unsigned short Pst[4][16][32];  // per-wave P staging
  const int lane = threadIdx.x & 31;
  const int wave = threadIdx.x >> 5;
  const int lh = lane & 15;
  const int kg = lane >> 4;

  const int tile = blockIdx.x * 4 + wave;  // 0 .. B*H*(S/16)-1
  const int QT = SS / 16;
  const int qt = tile % QT;
  const int bh = tile / QT;
  const int h = bh % HH;
  const int b = bh / HH;
  const int q0 = qt * 16;

  const unsigned short* qbase = qkv16 + (size_t)b * SS * E3 + (size_t)h * DD;
  const unsigned short* kbase = qbase + EE;
  const unsigned short* vbase = qbase + 2 * EE;

  // Q fragments: 16x64 -> two 16x32 A-fragments, each two b128 loads.
  Frag qa[2];
#pragma unroll
  for (int f = 0; f < 2; ++f) {
    const unsigned short* qr = qbase + (size_t)(q0 + lh) * E3 + f * 32 + 8 * kg;
    qa[f].q[0] = *reinterpret_cast<const uint32x4*>(qr);       // d = f*32+8kg .. +7
    qa[f].q[1] = *reinterpret_cast<const uint32x4*>(qr + 16);  // d = f*32+16+8kg .. +7
  }

  v8f o[4];
#pragma unroll
  for (int f = 0; f < 4; ++f) o[f] = vzero8();
  float mrow[8], lrow[8];
#pragma unroll
  for (int i = 0; i < 8; ++i) {
    mrow[i] = -3.0e38f;
    lrow[i] = 0.0f;
  }

  const int ntiles = (q0 + 16 + 31) / 32;  // causal: keys <= q0+15
  for (int t = 0; t < ntiles; ++t) {
    const int k0 = t * 32;

    // ---- stage V tile transposed: lane owns key k0+lane, scatters 64 d's ----
    {
      const unsigned short* vr = vbase + (size_t)(k0 + lane) * E3;
#pragma unroll
      for (int j = 0; j < 8; ++j) {
        Q16 vv;
        vv.q = *reinterpret_cast<const uint32x4*>(vr + j * 8);
#pragma unroll
        for (int c = 0; c < 8; ++c) Vst[wave][j * 8 + c][lane] = vv.u[c];
      }
    }

    // ---- scores S = Q @ K^T : 16x32 as two 16x16 accumulators ----
    v8f sc[2];
    sc[0] = vzero8();
    sc[1] = vzero8();
#pragma unroll
    for (int hf = 0; hf < 2; ++hf) {
      const unsigned short* krow = kbase + (size_t)(k0 + hf * 16 + lh) * E3;
#pragma unroll
      for (int df = 0; df < 2; ++df) {
        Frag kb;  // B operand = K^T (d x key); d in {8kg..+7, 16+8kg..+7} per lane
        const unsigned short* kp = krow + df * 32 + 8 * kg;
        kb.q[0] = *reinterpret_cast<const uint32x4*>(kp);
        kb.q[1] = *reinterpret_cast<const uint32x4*>(kp + 16);
        sc[hf] = __builtin_amdgcn_wmma_f32_16x16x32_bf16(
            false, qa[df].v, false, kb.v, (short)0, sc[hf], false, false);
      }
    }

    // ---- causal mask + online softmax (rows span 16 lanes within a half) ----
    float alpha[8];
#pragma unroll
    for (int i = 0; i < 8; ++i) {
      const int row = i + 8 * kg;
      const int q = q0 + row;
      const int key0 = k0 + lh;
      const int key1 = k0 + 16 + lh;
      float a0 = (key0 <= q) ? sc[0][i] : -3.0e38f;
      float a1 = (key1 <= q) ? sc[1][i] : -3.0e38f;
      float mx = fmaxf(a0, a1);
#pragma unroll
      for (int off = 8; off > 0; off >>= 1) mx = fmaxf(mx, __shfl_xor(mx, off, 32));
      const float newm = fmaxf(mrow[i], mx);
      const float al = __expf(mrow[i] - newm);
      const float p0 = __expf(a0 - newm);
      const float p1 = __expf(a1 - newm);
      float rs = p0 + p1;
#pragma unroll
      for (int off = 8; off > 0; off >>= 1) rs += __shfl_xor(rs, off, 32);
      lrow[i] = lrow[i] * al + rs;
      mrow[i] = newm;
      alpha[i] = al;
      Pst[wave][row][lh] = f2bf16(p0);
      Pst[wave][row][16 + lh] = f2bf16(p1);
    }

    // ---- P relayout: C-layout -> A-fragment via per-wave LDS (b128 reads) ----
    Frag pa;
    pa.q[0] = *reinterpret_cast<const uint32x4*>(&Pst[wave][lh][8 * kg]);
    pa.q[1] = *reinterpret_cast<const uint32x4*>(&Pst[wave][lh][16 + 8 * kg]);

    // ---- O = O*alpha + P @ V ----
#pragma unroll
    for (int f = 0; f < 4; ++f) {
#pragma unroll
      for (int i = 0; i < 8; ++i) o[f][i] *= alpha[i];
      Frag vb;  // B 32x16: key contiguous per lane from transposed LDS tile
      vb.q[0] = *reinterpret_cast<const uint32x4*>(&Vst[wave][f * 16 + lh][8 * kg]);
      vb.q[1] = *reinterpret_cast<const uint32x4*>(&Vst[wave][f * 16 + lh][16 + 8 * kg]);
      o[f] = __builtin_amdgcn_wmma_f32_16x16x32_bf16(
          false, pa.v, false, vb.v, (short)0, o[f], false, false);
    }
  }

  // ---- normalize and store bf16 [B,S,E] with heads re-merged ----
#pragma unroll
  for (int f = 0; f < 4; ++f)
#pragma unroll
    for (int i = 0; i < 8; ++i) {
      const int row = i + 8 * kg;
      const int s = q0 + row;
      const int col = h * DD + f * 16 + lh;
      out16[((size_t)(b * SS + s)) * EE + col] = f2bf16(o[f][i] / lrow[i]);
    }
}

// ---------------------------------------------------------------------------
// Orchestration. Workspace (256B-aligned chunks):
//   bf16: h1[M*E] qkv[M*3E] attno[M*E] h2[M*E] fc[M*4E]
//         w_attn16[E*3E] w_aproj16[E*E] w_fc16[E*4E] w_fproj16[4E*E]
//   f32 : x1[M*E]
//   total ~ 90 MB.
// ---------------------------------------------------------------------------
extern "C" void kernel_launch(void* const* d_in, const int* in_sizes, int n_in,
                              void* d_out, int out_size, void* d_ws, size_t ws_size,
                              hipStream_t stream) {
  (void)in_sizes; (void)n_in; (void)out_size; (void)ws_size;
  const float* x       = (const float*)d_in[0];
  const float* ln1_g   = (const float*)d_in[1];
  const float* ln1_b   = (const float*)d_in[2];
  const float* w_attn  = (const float*)d_in[3];
  const float* b_attn  = (const float*)d_in[4];
  const float* w_aproj = (const float*)d_in[5];
  const float* b_aproj = (const float*)d_in[6];
  const float* ln2_g   = (const float*)d_in[7];
  const float* ln2_b   = (const float*)d_in[8];
  const float* w_fc    = (const float*)d_in[9];
  const float* b_fc    = (const float*)d_in[10];
  const float* w_fproj = (const float*)d_in[11];
  const float* b_fproj = (const float*)d_in[12];
  float* out = (float*)d_out;

  const size_t ME = (size_t)MM * EE;
  char* p = (char*)d_ws;
  auto alloc = [&](size_t bytes) -> char* {
    char* r = p;
    p += (bytes + 255) & ~(size_t)255;
    return r;
  };
  unsigned short* h1    = (unsigned short*)alloc(ME * 2);
  unsigned short* qkvb  = (unsigned short*)alloc(3 * ME * 2);
  unsigned short* attno = (unsigned short*)alloc(ME * 2);
  unsigned short* h2    = (unsigned short*)alloc(ME * 2);
  unsigned short* fcb   = (unsigned short*)alloc(4 * ME * 2);
  float*          x1    = (float*)alloc(ME * 4);
  unsigned short* wat16 = (unsigned short*)alloc((size_t)EE * E3 * 2);
  unsigned short* wap16 = (unsigned short*)alloc((size_t)EE * EE * 2);
  unsigned short* wfc16 = (unsigned short*)alloc((size_t)EE * 4 * EE * 2);
  unsigned short* wfp16 = (unsigned short*)alloc((size_t)4 * EE * EE * 2);

  // 0) one-shot weight conversions to bf16
  {
    int n4;
    n4 = EE * E3 / 4;
    cvt_bf16_kernel<<<(n4 + 255) / 256, 256, 0, stream>>>(w_attn, wat16, n4);
    n4 = EE * EE / 4;
    cvt_bf16_kernel<<<(n4 + 255) / 256, 256, 0, stream>>>(w_aproj, wap16, n4);
    n4 = EE * 4 * EE / 4;
    cvt_bf16_kernel<<<(n4 + 255) / 256, 256, 0, stream>>>(w_fc, wfc16, n4);
    cvt_bf16_kernel<<<(n4 + 255) / 256, 256, 0, stream>>>(w_fproj, wfp16, n4);
  }

  // 1) LN1 -> bf16
  ln_kernel<<<MM, 256, 0, stream>>>(x, ln1_g, ln1_b, h1);
  // 2) qkv(bf16) = h1 @ w_attn + b_attn   [4096 x 2304], K=768
  gemm_bf16_kernel<0, false, true><<<dim3(3 * EE / 128, MM / 128), 256, 0, stream>>>(
      h1, wat16, b_attn, nullptr, qkvb, MM, 3 * EE, EE);
  // 3) causal flash attention -> bf16
  attn_kernel<<<(BB * HH * (SS / 16)) / 4, 128, 0, stream>>>(qkvb, attno);
  // 4) x1(f32) = x + attno @ w_aproj + b_aproj
  gemm_bf16_kernel<0, true, false><<<dim3(EE / 128, MM / 128), 256, 0, stream>>>(
      attno, wap16, b_aproj, x, x1, MM, EE, EE);
  // 5) LN2 -> bf16
  ln_kernel<<<MM, 256, 0, stream>>>(x1, ln2_g, ln2_b, h2);
  // 6) fc(bf16) = gelu(h2 @ w_fc + b_fc)  [4096 x 3072], K=768
  gemm_bf16_kernel<1, false, true><<<dim3(4 * EE / 128, MM / 128), 256, 0, stream>>>(
      h2, wfc16, b_fc, nullptr, fcb, MM, 4 * EE, EE);
  // 7) out(f32) = x1 + fc @ w_fproj + b_fproj  [4096 x 768], K=3072
  gemm_bf16_kernel<0, true, false><<<dim3(EE / 128, MM / 128), 256, 0, stream>>>(
      fcb, wfp16, b_fproj, x1, out, MM, EE, 4 * EE);
}
